// ResDavenetVQ_4896262717885
// MI455X (gfx1250) — matrix operand order repaired and checked by reading, not compiled, not run on hardware
//
#include <hip/hip_runtime.h>
#include <hip/hip_bf16.h>

typedef __attribute__((ext_vector_type(16))) __bf16          bf16x16;
typedef __attribute__((ext_vector_type(16))) unsigned short  us16;
typedef __attribute__((ext_vector_type(8)))  unsigned short  us8;
typedef __attribute__((ext_vector_type(8)))  float           f32x8;
typedef __attribute__((ext_vector_type(4)))  unsigned int    u32x4;
typedef __attribute__((ext_vector_type(4)))  int             i32x4;
typedef __attribute__((ext_vector_type(8)))  int             i32x8;

#define EPSBN 1e-5f

#if defined(__has_builtin)
#if __has_builtin(__builtin_amdgcn_tensor_load_to_lds) && __has_builtin(__builtin_amdgcn_s_wait_tensorcnt)
#define USE_TDM 1
#endif
#endif

__device__ __forceinline__ unsigned short f2bfu(float f) {
    unsigned u = __builtin_bit_cast(unsigned, f);
    unsigned r = u + 0x7FFFu + ((u >> 16) & 1u);   // round-to-nearest-even
    return (unsigned short)(r >> 16);
}

#ifdef USE_TDM
// 1-D contiguous global->LDS copy via the Tensor Data Mover.
// elems = number of 2-byte elements (<= 65535). D# per ISA 8.3/8.4:
//   group0: count=1 | lds_addr | global_addr[56:0] | type=2
//   group1: data_size=1(2B); tensor_dim0=elems; tensor_dim1=1; tile_dim0=elems;
//           tile_dim1=1; tensor_dim0_stride=elems
__device__ __forceinline__ void tdm_copy_to_lds(unsigned lds_off, const void* gptr,
                                                unsigned elems) {
    unsigned long long ga = (unsigned long long)gptr;
    u32x4 g0;
    g0[0] = 1u;                                            // count=1, user mode
    g0[1] = lds_off;                                       // lds_addr (bytes)
    g0[2] = (unsigned)(ga & 0xFFFFFFFFu);                  // global_addr[31:0]
    g0[3] = (unsigned)((ga >> 32) & 0x01FFFFFFu) | (2u << 30);  // addr[56:32] | type=2
    i32x8 g1;
    g1[0] = (int)(1u << 16);                               // wg_mask=0, data_size=1 (2B)
    g1[1] = (int)((elems & 0xFFFFu) << 16);                // tensor_dim0[15:0]
    g1[2] = (int)((elems >> 16) | (1u << 16));             // tensor_dim0[31:16] | tensor_dim1=1
    g1[3] = (int)(elems << 16);                            // tile_dim0 (dim1[31:16]=0)
    g1[4] = 1;                                             // tile_dim1=1, tile_dim2=0
    g1[5] = (int)elems;                                    // tensor_dim0_stride[31:0]
    g1[6] = 0;
    g1[7] = 0;
    i32x4 z = {0, 0, 0, 0};
#if __clang_major__ >= 23
    i32x8 z8 = {0, 0, 0, 0, 0, 0, 0, 0};
    __builtin_amdgcn_tensor_load_to_lds(g0, g1, z, z, z8, 0);
#else
    __builtin_amdgcn_tensor_load_to_lds(g0, g1, z, z, 0);
#endif
}
#endif

// ---------------------------------------------------------------------------
// Generic 1xKW conv as implicit GEMM on WMMA bf16 (fp32 accumulate).
//   Y[n][co][wo] = act( bn( sum_{ci,kw} W[co][ci][kw] * X[n][ci][wo*stride+kw-pad] ) + R )
// WB: weights pre-converted to bf16 and tiled [ci/32][co][kw][ci%32] so the
//     per-block tile is one contiguous chunk (TDM-friendly).
// Block: 256 thr = 8 waves; tile 64(Cout) x 64(Wout); wave -> 16x32 (2 accs, shared A).
// ---------------------------------------------------------------------------
__global__ __launch_bounds__(256) void conv_wmma(
    const float* __restrict__ X, const unsigned short* __restrict__ WB,
    const float* __restrict__ bnb, const float* __restrict__ bng,
    const float* __restrict__ bnm, const float* __restrict__ bnv,
    const float* __restrict__ R, float* __restrict__ Y,
    int Cin, int Cout, int Win, int Wout,
    int stride, int KW, int pad, int relu, int raw)
{
    const int n    = blockIdx.z;
    const int co0  = blockIdx.y * 64;
    const int wo0  = blockIdx.x * 64;
    const int tid  = threadIdx.x;
    const int lane = tid & 31;
    const int wave = tid >> 5;
    const int subm = wave >> 1;          // 0..3 : Cout subtile (x16)
    const int subn = wave & 1;           // 0..1 : Wout subtile (x32, two 16-wide halves)
    const int half = lane >> 4;          // hi/lo lane group
    const int m    = lane & 15;

    __shared__ unsigned short Wlds[64 * 9 * 32];   // [co][kw][ci] contiguous tile copy
    __shared__ unsigned short Xlds[136 * 40];      // [wl][ci] transposed, row stride 40

    const int span  = 64 * stride + KW - 1;        // input halo span (<= 136)
    const int wbase = wo0 * stride - pad;
    const int wtile_elems = 64 * KW * 32;

    f32x8 acc0 = {}, acc1 = {};

    for (int ci0 = 0; ci0 < Cin; ci0 += 32) {
        // ---- stage weight tile: contiguous bf16 chunk ----
        const unsigned short* wtile =
            WB + ((size_t)(ci0 >> 5) * Cout + co0) * (size_t)(KW * 32);
#ifdef USE_TDM
        if (wave == 0) {
            tdm_copy_to_lds((unsigned)(size_t)(void*)Wlds, wtile,
                            (unsigned)wtile_elems);
            __builtin_amdgcn_s_wait_tensorcnt(0);
        }
#else
        for (int idx = tid; idx < wtile_elems / 8; idx += 256)
            ((us8*)Wlds)[idx] = ((const us8*)wtile)[idx];
#endif
        // ---- prefetch next activation chunk ----
        if (ci0 + 32 < Cin) {
            int w0 = wbase < 0 ? 0 : wbase;
            __builtin_prefetch(&X[((size_t)n * Cin + (ci0 + 32 + lane)) * (size_t)Win + w0], 0, 1);
        }
        // ---- stage input tile (bf16, transposed) ----
        for (int idx = tid; idx < 32 * span; idx += 256) {
            int cil = idx / span, wl = idx - cil * span;
            int w = wbase + wl;
            float v = 0.f;
            if (w >= 0 && w < Win)
                v = X[((size_t)n * Cin + (ci0 + cil)) * (size_t)Win + w];
            Xlds[wl * 40 + cil] = f2bfu(v);
        }
        __syncthreads();

        for (int kw = 0; kw < KW; ++kw) {
            // A fragment (shared by both accs): 16x32 bf16.
            // lane<16: K 0..15 ; lane>=16: K 16..31
            const unsigned short* wrow = &Wlds[((subm * 16 + m) * KW + kw) * 32];
            us8 alo = *(const us8*)(wrow + 16 * half);
            us8 ahi = *(const us8*)(wrow + 16 * half + 8);
            us16 av;
#pragma unroll
            for (int e = 0; e < 8; ++e) { av[e] = alo[e]; av[e + 8] = ahi[e]; }
            bf16x16 A = __builtin_bit_cast(bf16x16, av);

            // B fragments: 32x16 bf16.  N = lane&15, K = e + 16*half (contiguous row)
            int wl0 = (subn * 32 + m) * stride + kw;
            int wl1 = (subn * 32 + 16 + m) * stride + kw;
            const unsigned short* xr0 = &Xlds[wl0 * 40 + 16 * half];
            const unsigned short* xr1 = &Xlds[wl1 * 40 + 16 * half];
            us8 b0l = *(const us8*)(xr0);
            us8 b0h = *(const us8*)(xr0 + 8);
            us8 b1l = *(const us8*)(xr1);
            us8 b1h = *(const us8*)(xr1 + 8);
            us16 bv0, bv1;
#pragma unroll
            for (int e = 0; e < 8; ++e) {
                bv0[e] = b0l[e]; bv0[e + 8] = b0h[e];
                bv1[e] = b1l[e]; bv1[e + 8] = b1h[e];
            }
            acc0 = __builtin_amdgcn_wmma_f32_16x16x32_bf16(
                     false, A, false, __builtin_bit_cast(bf16x16, bv0),
                     (short)0, acc0, false, false);
            acc1 = __builtin_amdgcn_wmma_f32_16x16x32_bf16(
                     false, A, false, __builtin_bit_cast(bf16x16, bv1),
                     (short)0, acc1, false, false);
        }
        __syncthreads();
    }

    // ---- epilogue: BN + residual + ReLU.  C/D layout: M = v + 8*half, N = lane&15
#pragma unroll
    for (int t = 0; t < 2; ++t) {
        const f32x8& acc = t ? acc1 : acc0;
        int wo = wo0 + subn * 32 + t * 16 + m;
#pragma unroll
        for (int v = 0; v < 8; ++v) {
            int co = co0 + subm * 16 + v + 8 * half;
            float o = acc[v];
            if (!raw) {
                float s = bng[co] * rsqrtf(bnv[co] + EPSBN);
                o = (o - bnm[co]) * s + bnb[co];
            }
            if (R) o += R[((size_t)n * Cout + co) * (size_t)Wout + wo];
            if (relu) o = fmaxf(o, 0.f);
            Y[((size_t)n * Cout + co) * (size_t)Wout + wo] = o;
        }
    }
}

// ---------------------------------------------------------------------------
// Weight pre-pass: fp32 (Cout,Cin,KW) -> bf16 tiled [ci/32][co][kw][ci%32]
// ---------------------------------------------------------------------------
__global__ __launch_bounds__(256) void cvt_weights(
    const float* __restrict__ src, unsigned short* __restrict__ dst,
    int Cout, int Cin, int KW)
{
    size_t idx = (size_t)blockIdx.x * 256 + threadIdx.x;
    size_t total = (size_t)Cout * Cin * KW;
    if (idx >= total) return;
    int kw = (int)(idx % KW);
    size_t t = idx / KW;
    int ci = (int)(t % Cin);
    int co = (int)(t / Cin);
    dst[(((size_t)(ci >> 5) * Cout + co) * KW + kw) * 32 + (ci & 31)] = f2bfu(src[idx]);
}

// ---------------------------------------------------------------------------
// Stem conv: 40x1 over H collapses H -> 1.
// ---------------------------------------------------------------------------
__global__ __launch_bounds__(256) void conv40_bn_relu(
    const float* __restrict__ X, const float* __restrict__ Wt,
    const float* __restrict__ bnb, const float* __restrict__ bng,
    const float* __restrict__ bnm, const float* __restrict__ bnv,
    float* __restrict__ Y)
{
    const int w = blockIdx.x * 256 + threadIdx.x;   // 0..2047
    const int n = blockIdx.y;
    float xr[40];
#pragma unroll
    for (int h = 0; h < 40; ++h)
        xr[h] = X[((size_t)n * 40 + h) * 2048 + w];
    for (int co = 0; co < 128; ++co) {
        float a = 0.f;
#pragma unroll
        for (int h = 0; h < 40; ++h) a = fmaf(Wt[co * 40 + h], xr[h], a);
        float s = bng[co] * rsqrtf(bnv[co] + EPSBN);
        float o = (a - bnm[co]) * s + bnb[co];
        Y[((size_t)n * 128 + co) * 2048 + w] = fmaxf(o, 0.f);
    }
}

__global__ void emb_norms(const float* __restrict__ emb, float* __restrict__ en, int C)
{
    int j = blockIdx.x * 256 + threadIdx.x;
    if (j < 512) {
        float s = 0.f;
        for (int c = 0; c < C; ++c) { float t = emb[j * C + c]; s = fmaf(t, t, s); }
        en[j] = s;
    }
}

// ---------------------------------------------------------------------------
// VQ epilogue: one wave per position.  d_j = ||e_j||^2 - 2*S[j] (+||f||^2).
// ---------------------------------------------------------------------------
__global__ __launch_bounds__(256) void vq_finalize(
    const float* __restrict__ act, const float* __restrict__ scores,
    const float* __restrict__ emb, const float* __restrict__ en,
    float* __restrict__ qout, float* __restrict__ loss,
    float* __restrict__ onehot, int C, int W)
{
    const int lane = threadIdx.x & 31;
    const int pos  = blockIdx.x * 8 + (threadIdx.x >> 5);
    const int n = pos / W, w = pos - n * W;

    float best = 3.402823466e+38f;
    int bidx = 0;
    for (int j = lane; j < 512; j += 32) {
        float d = en[j] - 2.f * scores[((size_t)n * 512 + j) * (size_t)W + w];
        if (d < best) { best = d; bidx = j; }
    }
#pragma unroll
    for (int off = 16; off; off >>= 1) {
        float ob = __shfl_xor(best, off, 32);
        int   oi = __shfl_xor(bidx, off, 32);
        if (ob < best || (ob == best && oi < bidx)) { best = ob; bidx = oi; }
    }
    float fn = 0.f;
    for (int c = lane; c < C; c += 32) {
        float t = act[((size_t)n * C + c) * (size_t)W + w];
        fn = fmaf(t, t, fn);
    }
#pragma unroll
    for (int off = 16; off; off >>= 1) fn += __shfl_xor(fn, off, 32);

    if (lane == 0) loss[pos] = (fn + best) / (float)C;   // COMMIT_COST = 1.0
    for (int j = lane; j < 512; j += 32)
        onehot[(size_t)pos * 512 + j] = (j == bidx) ? 1.f : 0.f;
    for (int c = lane; c < C; c += 32)
        qout[((size_t)n * C + c) * (size_t)W + w] = emb[bidx * C + c];
}

// ---------------------------------------------------------------------------
// Host side
// ---------------------------------------------------------------------------
struct BN4 { const float *b, *g, *m, *v; };
struct Blk { BN4 bn1, bn2, dbn; const float *c1, *c2, *down; };

static void conv(hipStream_t s, const float* X, const unsigned short* WB, const BN4& bn,
                 const float* R, float* Y, int Cin, int Cout, int Win, int Wout,
                 int stride, int KW, int pad, int relu, int raw)
{
    dim3 grid(Wout / 64, Cout / 64, 32);
    conv_wmma<<<grid, dim3(256), 0, s>>>(X, WB, bn.b, bn.g, bn.m, bn.v, R, Y,
                                         Cin, Cout, Win, Wout, stride, KW, pad, relu, raw);
}

extern "C" void kernel_launch(void* const* d_in, const int* in_sizes, int n_in,
                              void* d_out, int out_size, void* d_ws, size_t ws_size,
                              hipStream_t stream)
{
    (void)in_sizes; (void)n_in; (void)out_size; (void)ws_size;
    int i = 0;
    auto nxt = [&]() { return (const float*)d_in[i++]; };

    // jax tree flatten order (sorted dict keys): params first, then x.
    BN4 stem_bn = { nxt(), nxt(), nxt(), nxt() };      // bn1: b,g,m,v
    const float* w_stem = nxt();                       // conv1 (128,1,40,1)
    const float* emb2f  = nxt();                       // (512,128)
    const float* emb3f  = nxt();                       // (512,256)

    Blk blk[4][2];
    for (int l = 0; l < 4; ++l)
        for (int b = 0; b < 2; ++b) {
            Blk& B = blk[l][b];
            B.bn1 = { nxt(), nxt(), nxt(), nxt() };
            B.bn2 = { nxt(), nxt(), nxt(), nxt() };
            B.c1 = nxt(); B.c2 = nxt();
            if (b == 0) { B.down = nxt(); B.dbn = { nxt(), nxt(), nxt(), nxt() }; }
            else        { B.down = nullptr; }
        }
    const float* x = nxt();                            // (32,40,2048)

    // outputs: x(32*1024*128), l2(32768), l3(16384), oh2(32768*512), oh3(16384*512)
    float* out     = (float*)d_out;
    float* out_x   = out;
    float* out_l2  = out + 4194304;
    float* out_l3  = out_l2 + 32768;
    float* out_oh2 = out_l3 + 16384;
    float* out_oh3 = out_oh2 + 16777216;

    // workspace layout (floats):
    float* ws = (float*)d_ws;
    float* P0 = ws;                       // 8388608
    float* P1 = ws + 8388608;             // 8388608
    float* P2 = ws + 16777216;            // 4194304 (temps never exceed 4.2M)
    float* P3 = ws + 20971520;            // 4194304
    float* scores = ws + 25165824;        // 16777216
    float* enorm  = ws + 41943040;        // 512
    unsigned short* wb = (unsigned short*)(ws + 41943552);   // bf16 weight pool

    // ---- weight pre-conversion (bf16, TDM-friendly tiling) ----
    unsigned short* wbcur = wb;
    auto cvt = [&](const float* src, int Cout, int Cin, int KW) {
        unsigned short* dst = wbcur;
        size_t total = (size_t)Cout * Cin * KW;
        cvt_weights<<<dim3((unsigned)((total + 255) / 256)), dim3(256), 0, stream>>>(
            src, dst, Cout, Cin, KW);
        wbcur += total;
        return (const unsigned short*)dst;
    };
    const int planesL[4] = { 128, 256, 512, 1024 };
    const unsigned short *wc1[4][2], *wc2[4][2], *wdn[4];
    {
        int C = 128;
        for (int l = 0; l < 4; ++l) {
            int p = planesL[l];
            wc1[l][0] = cvt(blk[l][0].c1, p, C, 9);
            wc2[l][0] = cvt(blk[l][0].c2, p, p, 9);
            wdn[l]    = cvt(blk[l][0].down, p, C, 1);
            wc1[l][1] = cvt(blk[l][1].c1, p, p, 9);
            wc2[l][1] = cvt(blk[l][1].c2, p, p, 9);
            C = p;
        }
    }
    const unsigned short* wemb2 = cvt(emb2f, 512, 128, 1);
    const unsigned short* wemb3 = cvt(emb3f, 512, 256, 1);

    // ---- stem ----
    conv40_bn_relu<<<dim3(8, 32), dim3(256), 0, stream>>>(
        x, w_stem, stem_bn.b, stem_bn.g, stem_bn.m, stem_bn.v, P0);

    int C = 128, W = 2048;
    float* cur = P0;

    for (int l = 0; l < 4; ++l) {
        int planes = planesL[l];
        int Wout = W / 2;
        Blk& b0 = blk[l][0];
        conv(stream, cur, wc1[l][0], b0.bn1, nullptr, P2, C, planes, W, Wout, 2, 9, 4, 1, 0);
        conv(stream, cur, wdn[l],    b0.dbn, nullptr, P3, C, planes, W, Wout, 2, 1, 0, 0, 0);
        float* nxtbuf = (cur == P0) ? P1 : P0;
        conv(stream, P2, wc2[l][0], b0.bn2, P3, nxtbuf, planes, planes, Wout, Wout, 1, 9, 4, 1, 0);
        cur = nxtbuf; C = planes; W = Wout;

        Blk& b1 = blk[l][1];
        conv(stream, cur, wc1[l][1], b1.bn1, nullptr, P2, C, C, W, W, 1, 9, 4, 1, 0);
        nxtbuf = (cur == P0) ? P1 : P0;
        float* outp = (l == 3) ? out_x : nxtbuf;
        conv(stream, P2, wc2[l][1], b1.bn2, cur, outp, C, C, W, W, 1, 9, 4, 1, 0);
        cur = nxtbuf;

        if (l == 0 || l == 1) {
            const float* embf = (l == 0) ? emb2f : emb3f;
            const unsigned short* embb = (l == 0) ? wemb2 : wemb3;
            float* lo = (l == 0) ? out_l2 : out_l3;
            float* oh = (l == 0) ? out_oh2 : out_oh3;
            emb_norms<<<dim3(2), dim3(256), 0, stream>>>(embf, enorm, C);
            BN4 dummy = { embf, embf, embf, embf };
            conv(stream, cur, embb, dummy, nullptr, scores, C, 512, W, W, 1, 1, 0, 0, 1);
            float* q = (cur == P0) ? P1 : P0;
            vq_finalize<<<dim3(32 * W / 8), dim3(256), 0, stream>>>(
                cur, scores, embf, enorm, q, lo, oh, C, W);
            cur = q;
        }
    }
}
